// GPT_33328946217525
// MI455X (gfx1250) — matrix operand order repaired and checked by reading, not compiled, not run on hardware
//
#include <hip/hip_runtime.h>
#include <hip/hip_bf16.h>
#include <math.h>

// Problem dims (fixed by the reference)
#define B_  4
#define T_  1024
#define D_  1024
#define H_  16
#define L_  8
#define V_  4096
#define HD_ 64

typedef __attribute__((ext_vector_type(16))) __bf16 v16bf;
typedef __attribute__((ext_vector_type(8)))  __bf16 v8bf;
typedef __attribute__((ext_vector_type(8)))  float  v8f;

// ---------------------------------------------------------------------------
// WMMA helpers (CDNA5 v_wmma_f32_16x16x32_bf16, wave32)
// A-matrix 16x32 (MxK) per-lane layout (ISA 7.12.2):
//   m = lane&15, hi = lane>>4
//   elems 0..7  = A[m][hi*8 .. hi*8+7]
//   elems 8..15 = A[m][16+hi*8 .. 16+hi*8+7]
// B-matrix 32x16 (KxN) per-lane layout, sourced from row-major W[n][k] (=B^T):
//   n = lane&15, hi = lane>>4
//   elems 0..7 = W[n][hi*8 .. hi*8+7], elems 8..15 = W[n][16+hi*8 .. 16+hi*8+7]
// C/D 16x16: reg j holds C[j + 8*hi][lane&15]
// ---------------------------------------------------------------------------

__device__ __forceinline__ v16bf frag_cat(v8bf lo, v8bf hh) {
  return __builtin_shufflevector(lo, hh, 0,1,2,3,4,5,6,7,8,9,10,11,12,13,14,15);
}

__device__ __forceinline__ v16bf load_frag_a(const __bf16* row, int hi) {
  v8bf lo = *(const v8bf*)(row + hi * 8);
  v8bf hh = *(const v8bf*)(row + 16 + hi * 8);
  return frag_cat(lo, hh);
}

__device__ __forceinline__ v16bf load_frag_bT(const __bf16* row, int hi) {
  v8bf lo = *(const v8bf*)(row + hi * 8);
  v8bf hh = *(const v8bf*)(row + 16 + hi * 8);
  return frag_cat(lo, hh);
}

__device__ __forceinline__ v8f wmma_bf16(v16bf a, v16bf b, v8f c) {
  return __builtin_amdgcn_wmma_f32_16x16x32_bf16(false, a, false, b, (short)0, c,
                                                 false, false);
}

// ---------------------------------------------------------------------------
// Embedding: x[b,t,:] = tok_emb[idx[b,t],:] + pos_emb[t,:]
// ---------------------------------------------------------------------------
__global__ __launch_bounds__(256) void embed_kernel(
    const int* __restrict__ idx, const float* __restrict__ tok,
    const float* __restrict__ pos, float* __restrict__ x) {
  int i  = blockIdx.x * 256 + threadIdx.x;   // < B*T*D
  int d  = i & (D_ - 1);
  int bt = i >> 10;                          // /D_
  int t  = bt & (T_ - 1);
  x[i] = tok[(size_t)idx[bt] * D_ + d] + pos[(size_t)t * D_ + d];
}

// ---------------------------------------------------------------------------
// LayerNorm: one block (256 thr = 8 waves) per row of D_=1024
// ---------------------------------------------------------------------------
__global__ __launch_bounds__(256) void ln_kernel(
    const float* __restrict__ x, const float* __restrict__ w,
    const float* __restrict__ b, float* __restrict__ out) {
  const int row = blockIdx.x;
  const float* xr = x + (size_t)row * D_;
  float s = 0.f, s2 = 0.f;
  for (int i = threadIdx.x; i < D_; i += 256) {
    float v = xr[i];
    s += v;
    s2 += v * v;
  }
#pragma unroll
  for (int m = 16; m >= 1; m >>= 1) {
    s  += __shfl_xor(s,  m, 32);
    s2 += __shfl_xor(s2, m, 32);
  }
  __shared__ float rs[8], rs2[8];
  int wv = threadIdx.x >> 5;
  if ((threadIdx.x & 31) == 0) { rs[wv] = s; rs2[wv] = s2; }
  __syncthreads();
  float ts = 0.f, ts2 = 0.f;
#pragma unroll
  for (int i = 0; i < 8; ++i) { ts += rs[i]; ts2 += rs2[i]; }
  float mean = ts * (1.0f / D_);
  float var  = ts2 * (1.0f / D_) - mean * mean;
  float rstd = rsqrtf(var + 1e-5f);
  float* orow = out + (size_t)row * D_;
  for (int i = threadIdx.x; i < D_; i += 256)
    orow[i] = (xr[i] - mean) * rstd * w[i] + b[i];
}

// ---------------------------------------------------------------------------
// GEMM: C[M,N] = act( A[M,K] * W[N,K]^T + bias ) (+ resid)
// block = 256 thr (8 waves), tile 128x128, K-step 32, bf16 WMMA, f32 accum.
// Each wave owns a 32x64 sub-tile: 2 A-frags x 4 B-frags -> 8 WMMAs / K-step.
// ---------------------------------------------------------------------------
#define GBM 128
#define GBN 128
#define GBK 32
#define GLD 40  // bf16 row pitch (80B = 20 banks -> conflict free b128 reads)

template <bool BIAS, bool RESID, bool GELU>
__global__ __launch_bounds__(256) void gemm_bf16(
    const float* __restrict__ A, const float* __restrict__ W,
    const float* __restrict__ bias, const float* __restrict__ resid,
    float* __restrict__ C, int M, int N, int K) {
  __shared__ __align__(16) __bf16 As[GBM][GLD];
  __shared__ __align__(16) __bf16 Ws[GBN][GLD];

  const int tid   = threadIdx.x;
  const int lane  = tid & 31;
  const int wave  = tid >> 5;
  const int hi    = lane >> 4;
  const int lrow  = lane & 15;
  const int mbase = blockIdx.y * GBM;
  const int nbase = blockIdx.x * GBN;
  const int wm    = (wave >> 1) * 32;  // wave's m offset in tile (0,32,64,96)
  const int wn    = (wave & 1) * 64;   // wave's n offset in tile (0,64)

  const v8f vzero = {};
  v8f acc[2][4];
#pragma unroll
  for (int mt = 0; mt < 2; ++mt)
#pragma unroll
    for (int nt = 0; nt < 4; ++nt) acc[mt][nt] = vzero;

  // staging geometry: both tiles are 128x32 = 4096 floats, 16 per thread
  const int srow = tid >> 1;
  const int scol = (tid & 1) * 16;

  for (int k0 = 0; k0 < K; k0 += GBK) {
    {  // stage A tile
      const float* src = A + (size_t)(mbase + srow) * K + k0 + scol;
      __builtin_prefetch(src + GBK, 0, 0);  // next K-tile -> global_prefetch
#pragma unroll
      for (int i = 0; i < 16; i += 4) {
        float4 f = *(const float4*)(src + i);
        As[srow][scol + i + 0] = (__bf16)f.x;
        As[srow][scol + i + 1] = (__bf16)f.y;
        As[srow][scol + i + 2] = (__bf16)f.z;
        As[srow][scol + i + 3] = (__bf16)f.w;
      }
    }
    {  // stage W tile
      const float* src = W + (size_t)(nbase + srow) * K + k0 + scol;
      __builtin_prefetch(src + GBK, 0, 0);
#pragma unroll
      for (int i = 0; i < 16; i += 4) {
        float4 f = *(const float4*)(src + i);
        Ws[srow][scol + i + 0] = (__bf16)f.x;
        Ws[srow][scol + i + 1] = (__bf16)f.y;
        Ws[srow][scol + i + 2] = (__bf16)f.z;
        Ws[srow][scol + i + 3] = (__bf16)f.w;
      }
    }
    __syncthreads();

    v16bf af[2], bfr[4];
    af[0] = load_frag_a(&As[wm + lrow][0], hi);
    af[1] = load_frag_a(&As[wm + 16 + lrow][0], hi);
#pragma unroll
    for (int nt = 0; nt < 4; ++nt)
      bfr[nt] = load_frag_bT(&Ws[wn + nt * 16 + lrow][0], hi);
#pragma unroll
    for (int mt = 0; mt < 2; ++mt)
#pragma unroll
      for (int nt = 0; nt < 4; ++nt)
        acc[mt][nt] = wmma_bf16(af[mt], bfr[nt], acc[mt][nt]);
    __syncthreads();
  }

#pragma unroll
  for (int mt = 0; mt < 2; ++mt) {
#pragma unroll
    for (int nt = 0; nt < 4; ++nt) {
      int gn   = nbase + wn + nt * 16 + lrow;
      float bv = BIAS ? bias[gn] : 0.0f;
      v8f a = acc[mt][nt];
#pragma unroll
      for (int j = 0; j < 8; ++j) {
        int gm  = mbase + wm + mt * 16 + j + 8 * hi;
        float c = a[j] + bv;
        if (RESID) c += resid[(size_t)gm * N + gn];
        if (GELU)  c = 0.5f * c * (1.0f + erff(c * 0.70710678118654752f));
        C[(size_t)gm * N + gn] = c;
      }
    }
  }
}

// ---------------------------------------------------------------------------
// Flash attention: block = (b, h, 64-row Q tile), 128 thr (4 waves).
// Each wave owns 16 Q rows; streams 64-wide KV tiles with online softmax.
// ---------------------------------------------------------------------------
#define APITCH 72  // bf16 pitch (144B, mult of 16B)

__global__ __launch_bounds__(128) void attn_kernel(
    const float* __restrict__ q, const float* __restrict__ k,
    const float* __restrict__ v, float* __restrict__ y) {
  __shared__ __align__(16) __bf16 Qs[64][APITCH];
  __shared__ __align__(16) __bf16 Ks[64][APITCH];
  __shared__ __align__(16) __bf16 Vt[64][APITCH];      // transposed: Vt[d][key]
  __shared__ __align__(16) __bf16 Ps[4][16][APITCH];   // per-wave P tile

  const int qt   = blockIdx.x;  // q tile 0..15
  const int hh   = blockIdx.y;  // head
  const int bb   = blockIdx.z;  // batch
  const int tid  = threadIdx.x;
  const int lane = tid & 31;
  const int wv   = tid >> 5;    // wave 0..3
  const int hi   = lane >> 4;
  const int lrow = lane & 15;

  const size_t headoff = (size_t)bb * T_ * D_ + (size_t)hh * HD_;
  const float scale = 0.125f;  // 1/sqrt(HD_)

  {  // stage Q tile (pre-scaled): 32 floats / thread
    int row = tid >> 1;
    int col = (tid & 1) * 32;
    const float* src = q + headoff + (size_t)(qt * 64 + row) * D_ + col;
#pragma unroll
    for (int i = 0; i < 32; i += 4) {
      float4 f = *(const float4*)(src + i);
      Qs[row][col + i + 0] = (__bf16)(f.x * scale);
      Qs[row][col + i + 1] = (__bf16)(f.y * scale);
      Qs[row][col + i + 2] = (__bf16)(f.z * scale);
      Qs[row][col + i + 3] = (__bf16)(f.w * scale);
    }
  }

  v8f o0 = {}, o1 = {}, o2 = {}, o3 = {};
  float mrun[8], lrun[8];
#pragma unroll
  for (int j = 0; j < 8; ++j) { mrun[j] = -1e30f; lrun[j] = 0.f; }

  for (int kt = 0; kt <= qt; ++kt) {
    __syncthreads();  // (A) prior readers of Ks/Vt done; Qs visible (iter 0)
    {                 // stage K tile + transposed V tile
      int row = tid >> 1;
      int col = (tid & 1) * 32;
      const float* srck = k + headoff + (size_t)(kt * 64 + row) * D_ + col;
      const float* srcv = v + headoff + (size_t)(kt * 64 + row) * D_ + col;
      __builtin_prefetch(srck + (size_t)64 * D_, 0, 0);  // next KV tile
      __builtin_prefetch(srcv + (size_t)64 * D_, 0, 0);
#pragma unroll
      for (int i = 0; i < 32; i += 4) {
        float4 fk = *(const float4*)(srck + i);
        Ks[row][col + i + 0] = (__bf16)fk.x;
        Ks[row][col + i + 1] = (__bf16)fk.y;
        Ks[row][col + i + 2] = (__bf16)fk.z;
        Ks[row][col + i + 3] = (__bf16)fk.w;
        float4 fv = *(const float4*)(srcv + i);
        Vt[col + i + 0][row] = (__bf16)fv.x;
        Vt[col + i + 1][row] = (__bf16)fv.y;
        Vt[col + i + 2][row] = (__bf16)fv.z;
        Vt[col + i + 3][row] = (__bf16)fv.w;
      }
    }
    __syncthreads();  // (B)

    // S = (Q*scale) . K^T   : 16x64 per wave
    v8f s0 = {}, s1 = {}, s2 = {}, s3 = {};
    const __bf16* qrow = &Qs[wv * 16 + lrow][0];
#pragma unroll
    for (int kk = 0; kk < 2; ++kk) {
      v16bf aq = load_frag_a(qrow + kk * 32, hi);
      s0 = wmma_bf16(aq, load_frag_bT(&Ks[0  + lrow][kk * 32], hi), s0);
      s1 = wmma_bf16(aq, load_frag_bT(&Ks[16 + lrow][kk * 32], hi), s1);
      s2 = wmma_bf16(aq, load_frag_bT(&Ks[32 + lrow][kk * 32], hi), s2);
      s3 = wmma_bf16(aq, load_frag_bT(&Ks[48 + lrow][kk * 32], hi), s3);
    }

    if (kt == qt) {  // causal mask, diagonal tile only
      const int qb = qt * 64 + wv * 16;
      const int kb = kt * 64 + lrow;
#pragma unroll
      for (int j = 0; j < 8; ++j) {
        int qr = qb + j + 8 * hi;
        if (kb +  0 > qr) s0[j] = -1e30f;
        if (kb + 16 > qr) s1[j] = -1e30f;
        if (kb + 32 > qr) s2[j] = -1e30f;
        if (kb + 48 > qr) s3[j] = -1e30f;
      }
    }

    // online softmax (xor masks <16 keep the two 16-lane halves separate)
#pragma unroll
    for (int j = 0; j < 8; ++j) {
      float mx = fmaxf(fmaxf(s0[j], s1[j]), fmaxf(s2[j], s3[j]));
#pragma unroll
      for (int m = 8; m >= 1; m >>= 1) mx = fmaxf(mx, __shfl_xor(mx, m, 32));
      float mnew  = fmaxf(mrun[j], mx);
      float alpha = __expf(mrun[j] - mnew);
      mrun[j] = mnew;
      float p0 = __expf(s0[j] - mnew);
      float p1 = __expf(s1[j] - mnew);
      float p2 = __expf(s2[j] - mnew);
      float p3 = __expf(s3[j] - mnew);
      float rs = p0 + p1 + p2 + p3;
#pragma unroll
      for (int m = 8; m >= 1; m >>= 1) rs += __shfl_xor(rs, m, 32);
      lrun[j] = lrun[j] * alpha + rs;
      o0[j] *= alpha; o1[j] *= alpha; o2[j] *= alpha; o3[j] *= alpha;
      int pr = j + 8 * hi;  // C-layout row -> A-layout row via LDS
      Ps[wv][pr][lrow +  0] = (__bf16)p0;
      Ps[wv][pr][lrow + 16] = (__bf16)p1;
      Ps[wv][pr][lrow + 32] = (__bf16)p2;
      Ps[wv][pr][lrow + 48] = (__bf16)p3;
    }
    __syncthreads();  // (C)

    // O += P . V
    const __bf16* prow = &Ps[wv][lrow][0];
#pragma unroll
    for (int kk = 0; kk < 2; ++kk) {
      v16bf ap = load_frag_a(prow + kk * 32, hi);
      o0 = wmma_bf16(ap, load_frag_bT(&Vt[0  + lrow][kk * 32], hi), o0);
      o1 = wmma_bf16(ap, load_frag_bT(&Vt[16 + lrow][kk * 32], hi), o1);
      o2 = wmma_bf16(ap, load_frag_bT(&Vt[32 + lrow][kk * 32], hi), o2);
      o3 = wmma_bf16(ap, load_frag_bT(&Vt[48 + lrow][kk * 32], hi), o3);
    }
  }

#pragma unroll
  for (int j = 0; j < 8; ++j) {
    float inv = 1.0f / lrun[j];
    int qr = qt * 64 + wv * 16 + j + 8 * hi;
    float* dst = y + ((size_t)bb * T_ + qr) * D_ + (size_t)hh * HD_;
    dst[ 0 + lrow] = o0[j] * inv;
    dst[16 + lrow] = o1[j] * inv;
    dst[32 + lrow] = o2[j] * inv;
    dst[48 + lrow] = o3[j] * inv;
  }
}

// ---------------------------------------------------------------------------
// Host orchestration (all on `stream`, graph-capture safe)
// ---------------------------------------------------------------------------
extern "C" void kernel_launch(void* const* d_in, const int* in_sizes, int n_in,
                              void* d_out, int out_size, void* d_ws,
                              size_t ws_size, hipStream_t stream) {
  (void)in_sizes; (void)n_in; (void)out_size; (void)ws_size;
  const int*   idx  = (const int*)  d_in[0];
  const float* tok  = (const float*)d_in[1];
  const float* pos  = (const float*)d_in[2];
  const float* ln1w = (const float*)d_in[3];
  const float* ln1b = (const float*)d_in[4];
  const float* Wq   = (const float*)d_in[5];
  const float* bq   = (const float*)d_in[6];
  const float* Wk   = (const float*)d_in[7];
  const float* bk   = (const float*)d_in[8];
  const float* Wv   = (const float*)d_in[9];
  const float* bv   = (const float*)d_in[10];
  const float* Wp   = (const float*)d_in[11];
  const float* bp   = (const float*)d_in[12];
  const float* ln2w = (const float*)d_in[13];
  const float* ln2b = (const float*)d_in[14];
  const float* W1   = (const float*)d_in[15];
  const float* b1   = (const float*)d_in[16];
  const float* W2   = (const float*)d_in[17];
  const float* b2   = (const float*)d_in[18];
  const float* lnfw = (const float*)d_in[19];
  const float* lnfb = (const float*)d_in[20];
  const float* Wlog = (const float*)d_in[21];
  const float* Wdev = (const float*)d_in[22];

  const int M = B_ * T_;              // 4096 rows
  const size_t SZ = (size_t)M * D_;   // 4M floats
  float* x   = (float*)d_ws;          // [M,D]
  float* h   = x + SZ;                // [M,D]
  float* qb  = h + SZ;                // [M,D]
  float* kb  = qb + SZ;               // [M,D]
  float* vb  = kb + SZ;               // [M,D]
  float* yb  = vb + SZ;               // [M,D]
  float* hid = qb;                    // [M,4D] aliases q/k/v/y (dead by then)

  embed_kernel<<<dim3(M * D_ / 256), dim3(256), 0, stream>>>(idx, tok, pos, x);

  const dim3 gD(D_ / GBN, M / GBM);      // N=1024 gemms
  const dim3 g4D(4 * D_ / GBN, M / GBM); // N=4096 gemms
  const dim3 gAttn(T_ / 64, H_, B_);

  for (int l = 0; l < L_; ++l) {
    const size_t oDD = (size_t)l * D_ * D_;
    const size_t o4  = (size_t)l * 4 * D_ * D_;
    ln_kernel<<<dim3(M), dim3(256), 0, stream>>>(x, ln1w + l * D_, ln1b + l * D_, h);
    gemm_bf16<true, false, false><<<gD, 256, 0, stream>>>(
        h, Wq + oDD, bq + l * D_, nullptr, qb, M, D_, D_);
    gemm_bf16<true, false, false><<<gD, 256, 0, stream>>>(
        h, Wk + oDD, bk + l * D_, nullptr, kb, M, D_, D_);
    gemm_bf16<true, false, false><<<gD, 256, 0, stream>>>(
        h, Wv + oDD, bv + l * D_, nullptr, vb, M, D_, D_);
    attn_kernel<<<gAttn, 128, 0, stream>>>(qb, kb, vb, yb);
    gemm_bf16<true, true, false><<<gD, 256, 0, stream>>>(
        yb, Wp + oDD, bp + l * D_, x, x, M, D_, D_);
    ln_kernel<<<dim3(M), dim3(256), 0, stream>>>(x, ln2w + l * D_, ln2b + l * D_, h);
    gemm_bf16<true, false, true><<<g4D, 256, 0, stream>>>(
        h, W1 + o4, b1 + (size_t)l * 4 * D_, nullptr, hid, M, 4 * D_, D_);
    gemm_bf16<true, true, false><<<gD, 256, 0, stream>>>(
        hid, W2 + o4, b2 + l * D_, x, x, M, D_, 4 * D_);
  }

  ln_kernel<<<dim3(M), dim3(256), 0, stream>>>(x, lnfw, lnfb, h);
  float* out = (float*)d_out;
  const dim3 gV(V_ / GBN, M / GBM);
  gemm_bf16<false, false, false><<<gV, 256, 0, stream>>>(
      h, Wlog, nullptr, nullptr, out, M, V_, D_);
  gemm_bf16<false, false, false><<<gV, 256, 0, stream>>>(
      h, Wdev, nullptr, nullptr, out + (size_t)M * V_, M, V_, D_);
}
